// LinearAttention_pos_1_75041668596132
// MI455X (gfx1250) — matrix operand order repaired and checked by reading, not compiled, not run on hardware
//
#include <hip/hip_runtime.h>
#include <hip/hip_bf16.h>
#include <cstdint>

// ---------------------------------------------------------------------------
// CDNA5 (gfx1250) wave32 WMMA types
// ---------------------------------------------------------------------------
typedef __attribute__((ext_vector_type(16))) __bf16 v16bf;
typedef __attribute__((ext_vector_type(8)))  float  v8f;

#define NTOK   196      // H*W = 14*14
#define NTIL   13       // ceil(196/16)
#define DH     64       // dim_head
#define HEADS  8

union FragBF {           // one WMMA 16x32 bf16 operand (8 VGPRs)
  v16bf        v;
  unsigned int u[8];     // u[r] = packed bf16 pair (K = 2r, 2r+1 of the lane's slice)
};

__device__ __forceinline__ unsigned short f2bf(float f) {
  unsigned int u = __float_as_uint(f);             // RNE f32 -> bf16 bits
  u += 0x7fffu + ((u >> 16) & 1u);
  return (unsigned short)(u >> 16);
}
__device__ __forceinline__ unsigned int pack2bf(float lo, float hi) {
  return (unsigned int)f2bf(lo) | ((unsigned int)f2bf(hi) << 16);
}
// A-matrix (16x32 bf16): pair r (r=0..7) K-offset within the 32-wide k-slice
__device__ __forceinline__ int apair(int r, int half) {
  return half * 16 + 2 * r;
}
// B-matrix (32x16 bf16): pair r K-offset = half*16 + 2r

// ---------------------------------------------------------------------------
// Kernel 1: depthwise 3x3 on x [B,256,14,14] f32 -> dwx_t [B,196,256] bf16
// (transposed so the pw-qkv GEMM B-fragments are contiguous along K=c)
// ---------------------------------------------------------------------------
__global__ __launch_bounds__(256)
void dw3x3_x_kernel(const float* __restrict__ x, const float* __restrict__ w,
                    unsigned short* __restrict__ out, int total) {
  int idx = blockIdx.x * 256 + threadIdx.x;
  if (idx >= total) return;
  int c    = idx % 256;          // fastest: coalesced transposed store
  int rest = idx / 256;
  int p    = rest % NTOK;
  int b    = rest / NTOK;
  int y = p / 14, xx = p % 14;
  const float* wp = w + c * 9;
  const float* in = x + ((size_t)b * 256 + c) * NTOK;
  float s = 0.f;
#pragma unroll
  for (int dy = -1; dy <= 1; ++dy)
#pragma unroll
    for (int dx = -1; dx <= 1; ++dx) {
      int yy = y + dy, xc = xx + dx;
      if (yy >= 0 && yy < 14 && xc >= 0 && xc < 14)
        s += in[yy * 14 + xc] * wp[(dy + 1) * 3 + (dx + 1)];
    }
  out[((size_t)b * NTOK + p) * 256 + c] = f2bf(s);
}

// ---------------------------------------------------------------------------
// Kernel 2: pointwise qkv GEMM.  qkv[o,p] = sum_c W[o,c]*dwx[p,c], o=1536,c=256
// q,k stored [B,h,n,d]; v stored transposed [B,h,d,n] (for attn@v B-frags).
// grid = B*12, 256 thr (8 waves); wave -> one 16-row o-tile, 13 n-tiles.
// ---------------------------------------------------------------------------
__global__ __launch_bounds__(256)
void pw_qkv_kernel(const float* __restrict__ W,            // [1536,256]
                   const unsigned short* __restrict__ dwx, // [B,196,256] (transposed)
                   unsigned short* __restrict__ q_s,
                   unsigned short* __restrict__ k_s,
                   unsigned short* __restrict__ v_t) {
  int b      = blockIdx.x / 12;
  int mchunk = blockIdx.x % 12;
  const unsigned short* xin = dwx + (size_t)b * NTOK * 256;
  int wave = threadIdx.x >> 5, lane = threadIdx.x & 31;
  int half = lane >> 4, lcol = lane & 15;
  int mtile = mchunk * 8 + wave;           // 0..95
  int obase = mtile * 16;
  int arow  = obase + lcol;                // A row (o)

  v8f acc[NTIL] = {};
  for (int kk = 0; kk < 256; kk += 32) {
    FragBF a;
    const float* wrow = W + (size_t)arow * 256 + kk;
#pragma unroll
    for (int r = 0; r < 8; ++r) {
      int off = apair(r, half);
      a.u[r] = pack2bf(wrow[off], wrow[off + 1]);
    }
#pragma unroll
    for (int nt = 0; nt < NTIL; ++nt) {
      int p  = nt * 16 + lcol;
      int pc = p < NTOK ? p : NTOK - 1;
      const unsigned short* xr = xin + (size_t)pc * 256 + kk + half * 16;
      FragBF bb;
#pragma unroll
      for (int r = 0; r < 8; ++r)
        bb.u[r] = *(const unsigned int*)(xr + 2 * r);
      acc[nt] = __builtin_amdgcn_wmma_f32_16x16x32_bf16(
          false, a.v, false, bb.v, (short)0, acc[nt], false, false);
    }
  }
#pragma unroll
  for (int nt = 0; nt < NTIL; ++nt) {
    int p = nt * 16 + lcol;
    if (p >= NTOK) continue;
#pragma unroll
    for (int r = 0; r < 8; ++r) {
      int o   = obase + half * 8 + r;
      int sec = o >> 9;
      int oc  = o & 511;
      int d = oc >> 3, h = oc & 7;
      unsigned short val = f2bf(acc[nt][r]);
      if (sec == 2)      // v transposed: [B,h,d,n]
        v_t[(((size_t)b * HEADS + h) * DH + d) * NTOK + p] = val;
      else {
        unsigned short* dst = (sec == 0) ? q_s : k_s;
        dst[(((size_t)b * HEADS + h) * NTOK + p) * DH + d] = val;
      }
    }
  }
}

// ---------------------------------------------------------------------------
// Kernel 3: scores = (q k^T + bias) * 0.125, row softmax -> attn (f32, d_out)
// grid = B*8; 8 waves, wave handles row-tiles {w, w+8}.
// pos_table (729x8 f32, 23KB) cached in LDS.
// ---------------------------------------------------------------------------
__global__ __launch_bounds__(256)
void attn_scores_softmax_kernel(const unsigned short* __restrict__ q_s,
                                const unsigned short* __restrict__ k_s,
                                const float* __restrict__ pos_table,
                                float* __restrict__ attn) {
  __shared__ float pos_lds[729 * HEADS];
  for (int i = threadIdx.x; i < 729 * HEADS; i += 256) pos_lds[i] = pos_table[i];
  __syncthreads();

  int bh = blockIdx.x;
  int h  = bh & (HEADS - 1);
  const unsigned short* q = q_s + (size_t)bh * NTOK * DH;
  const unsigned short* k = k_s + (size_t)bh * NTOK * DH;
  float* at = attn + (size_t)bh * NTOK * NTOK;

  int wave = threadIdx.x >> 5, lane = threadIdx.x & 31;
  int half = lane >> 4, lcol = lane & 15;

  for (int mt = wave; mt < NTIL; mt += 8) {
    int arow  = mt * 16 + lcol;
    int arowc = arow < NTOK ? arow : NTOK - 1;
    const unsigned short* qrow = q + (size_t)arowc * DH;
    FragBF a0, a1;                      // K = 0..31 and 32..63
#pragma unroll
    for (int r = 0; r < 8; ++r) {
      int off = apair(r, half);
      a0.u[r] = *(const unsigned int*)(qrow + off);
      a1.u[r] = *(const unsigned int*)(qrow + 32 + off);
    }

    v8f acc[NTIL] = {};
#pragma unroll
    for (int nt = 0; nt < NTIL; ++nt) {
      int bcol  = nt * 16 + lcol;
      int bcolc = bcol < NTOK ? bcol : NTOK - 1;
      const unsigned short* krow = k + (size_t)bcolc * DH + half * 16;
      FragBF b0, b1;
#pragma unroll
      for (int r = 0; r < 8; ++r) {
        b0.u[r] = *(const unsigned int*)(krow + 2 * r);
        b1.u[r] = *(const unsigned int*)(krow + 32 + 2 * r);
      }
      acc[nt] = __builtin_amdgcn_wmma_f32_16x16x32_bf16(
          false, a0.v, false, b0.v, (short)0, acc[nt], false, false);
      acc[nt] = __builtin_amdgcn_wmma_f32_16x16x32_bf16(
          false, a1.v, false, b1.v, (short)0, acc[nt], false, false);
    }

    // bias + scale + mask  (C layout: VGPR r -> row mt*16+half*8+r, lane -> col)
#pragma unroll
    for (int nt = 0; nt < NTIL; ++nt) {
      int j = nt * 16 + lcol;
#pragma unroll
      for (int r = 0; r < 8; ++r) {
        int i = mt * 16 + half * 8 + r;
        float s;
        if (i < NTOK && j < NTOK) {
          int yi = i / 14, xi = i % 14, yj = j / 14, xj = j % 14;
          int rel = (yi - yj + 13) * 27 + (xi - xj + 13);
          s = (acc[nt][r] + pos_lds[rel * HEADS + h]) * 0.125f;
        } else {
          s = -3.0e38f;
        }
        acc[nt][r] = s;
      }
    }

    // softmax per row: reduce across the 16 lanes of this half via shfl_xor
    float rsum[8];
#pragma unroll
    for (int r = 0; r < 8; ++r) {
      float mx = -3.0e38f;
#pragma unroll
      for (int nt = 0; nt < NTIL; ++nt) mx = fmaxf(mx, acc[nt][r]);
#pragma unroll
      for (int o = 8; o >= 1; o >>= 1) mx = fmaxf(mx, __shfl_xor(mx, o, 32));
      float su = 0.f;
#pragma unroll
      for (int nt = 0; nt < NTIL; ++nt) {
        float e = __expf(acc[nt][r] - mx);   // masked entries -> exp(-huge)=0
        acc[nt][r] = e;
        su += e;
      }
#pragma unroll
      for (int o = 8; o >= 1; o >>= 1) su += __shfl_xor(su, o, 32);
      rsum[r] = 1.0f / su;
    }

#pragma unroll
    for (int nt = 0; nt < NTIL; ++nt) {
      int j = nt * 16 + lcol;
      if (j >= NTOK) continue;
#pragma unroll
      for (int r = 0; r < 8; ++r) {
        int i = mt * 16 + half * 8 + r;
        if (i < NTOK) at[(size_t)i * NTOK + j] = acc[nt][r] * rsum[r];
      }
    }
  }
}

// ---------------------------------------------------------------------------
// Kernel 4: out = attn @ v  -> out_s [B,h,196,64] f32.
// attn read back from d_out (157MB -> lives in 192MB L2). attn rows are
// contiguous so A-fragments are packed float-pair loads; v transposed so
// B-fragments are contiguous u32 pairs. K padded to 224, pair-masked (196 even).
// ---------------------------------------------------------------------------
__global__ __launch_bounds__(256)
void attn_v_kernel(const float* __restrict__ attn,
                   const unsigned short* __restrict__ v_t, // [B,h,64,196]
                   float* __restrict__ out_s) {
  int bh = blockIdx.x;
  const float* at = attn + (size_t)bh * NTOK * NTOK;
  const unsigned short* v = v_t + (size_t)bh * DH * NTOK;
  float* op = out_s + (size_t)bh * NTOK * DH;
  int wave = threadIdx.x >> 5, lane = threadIdx.x & 31;
  int half = lane >> 4, lcol = lane & 15;

  for (int mt = wave; mt < NTIL; mt += 8) {
    int arow  = mt * 16 + lcol;
    int arowc = arow < NTOK ? arow : NTOK - 1;
    const float* arp = at + (size_t)arowc * NTOK;
    v8f acc[4] = {};
    for (int kk = 0; kk < 224; kk += 32) {
      FragBF a;
#pragma unroll
      for (int r = 0; r < 8; ++r) {
        int kc = kk + apair(r, half);           // even
        float lo = (kc     < NTOK) ? arp[kc]     : 0.f;
        float hi = (kc + 1 < NTOK) ? arp[kc + 1] : 0.f;
        a.u[r] = pack2bf(lo, hi);
      }
#pragma unroll
      for (int nt = 0; nt < 4; ++nt) {
        int d = nt * 16 + lcol;
        const unsigned short* vrow = v + (size_t)d * NTOK;
        FragBF bb;
#pragma unroll
        for (int r = 0; r < 8; ++r) {
          int j = kk + half * 16 + 2 * r;       // even; NTOK even -> pair mask exact
          bb.u[r] = (j < NTOK) ? *(const unsigned int*)(vrow + j) : 0u;
        }
        acc[nt] = __builtin_amdgcn_wmma_f32_16x16x32_bf16(
            false, a.v, false, bb.v, (short)0, acc[nt], false, false);
      }
    }
#pragma unroll
    for (int nt = 0; nt < 4; ++nt) {
      int d = nt * 16 + lcol;
#pragma unroll
      for (int r = 0; r < 8; ++r) {
        int i = mt * 16 + half * 8 + r;
        if (i < NTOK) op[(size_t)i * DH + d] = acc[nt][r];
      }
    }
  }
}

// ---------------------------------------------------------------------------
// Kernel 5: depthwise 3x3 on rearranged attention output.
// input (b, ch, y, x) = out_s[b, ch%8, y*14+x, ch/8]; -> dwo_t [B,196,512] bf16
// ---------------------------------------------------------------------------
__global__ __launch_bounds__(256)
void dw3x3_out_kernel(const float* __restrict__ os, const float* __restrict__ w,
                      unsigned short* __restrict__ dwo, int total) {
  int idx = blockIdx.x * 256 + threadIdx.x;
  if (idx >= total) return;
  int ch   = idx % 512;          // fastest: coalesced transposed store
  int rest = idx / 512;
  int p    = rest % NTOK;
  int b    = rest / NTOK;
  int h = ch & 7, d = ch >> 3;
  int y = p / 14, xx = p % 14;
  const float* wp   = w + ch * 9;
  const float* base = os + (((size_t)b * HEADS + h) * NTOK) * DH + d;
  float s = 0.f;
#pragma unroll
  for (int dy = -1; dy <= 1; ++dy)
#pragma unroll
    for (int dx = -1; dx <= 1; ++dx) {
      int yy = y + dy, xc = xx + dx;
      if (yy >= 0 && yy < 14 && xc >= 0 && xc < 14)
        s += base[(size_t)(yy * 14 + xc) * DH] * wp[(dy + 1) * 3 + (dx + 1)];
    }
  dwo[((size_t)b * NTOK + p) * 512 + ch] = f2bf(s);
}

// ---------------------------------------------------------------------------
// Kernel 6: pointwise out GEMM. out[o,p] = sum_c Wout[o,c]*dwo[p,c], o=256,c=512
// grid = B*2; wave -> one 16-row o-tile.
// ---------------------------------------------------------------------------
__global__ __launch_bounds__(256)
void pw_out_kernel(const float* __restrict__ W,             // [256,512]
                   const unsigned short* __restrict__ dwo,  // [B,196,512] (transposed)
                   float* __restrict__ outp) {              // [B,256,196]
  int b  = blockIdx.x >> 1;
  int mh = blockIdx.x & 1;
  const unsigned short* xin = dwo + (size_t)b * NTOK * 512;
  int wave = threadIdx.x >> 5, lane = threadIdx.x & 31;
  int half = lane >> 4, lcol = lane & 15;
  int mtile = mh * 8 + wave;
  int obase = mtile * 16;
  int arow  = obase + lcol;                // < 256

  v8f acc[NTIL] = {};
  for (int kk = 0; kk < 512; kk += 32) {
    FragBF a;
    const float* wrow = W + (size_t)arow * 512 + kk;
#pragma unroll
    for (int r = 0; r < 8; ++r) {
      int off = apair(r, half);
      a.u[r] = pack2bf(wrow[off], wrow[off + 1]);
    }
#pragma unroll
    for (int nt = 0; nt < NTIL; ++nt) {
      int p  = nt * 16 + lcol;
      int pc = p < NTOK ? p : NTOK - 1;
      const unsigned short* xr = xin + (size_t)pc * 512 + kk + half * 16;
      FragBF bb;
#pragma unroll
      for (int r = 0; r < 8; ++r)
        bb.u[r] = *(const unsigned int*)(xr + 2 * r);
      acc[nt] = __builtin_amdgcn_wmma_f32_16x16x32_bf16(
          false, a.v, false, bb.v, (short)0, acc[nt], false, false);
    }
  }
#pragma unroll
  for (int nt = 0; nt < NTIL; ++nt) {
    int p = nt * 16 + lcol;
    if (p >= NTOK) continue;
#pragma unroll
    for (int r = 0; r < 8; ++r) {
      int o = obase + half * 8 + r;
      outp[((size_t)b * 256 + o) * NTOK + p] = acc[nt][r];
    }
  }
}

// ---------------------------------------------------------------------------
// Launch
// ---------------------------------------------------------------------------
extern "C" void kernel_launch(void* const* d_in, const int* in_sizes, int n_in,
                              void* d_out, int out_size, void* d_ws, size_t ws_size,
                              hipStream_t stream) {
  (void)n_in; (void)out_size; (void)ws_size;
  const float* x        = (const float*)d_in[0];  // [B,256,14,14]
  const float* w_dw_qkv = (const float*)d_in[1];  // [256,1,3,3]
  const float* w_pw_qkv = (const float*)d_in[2];  // [1536,256]
  const float* w_dw_out = (const float*)d_in[3];  // [512,1,3,3]
  const float* w_pw_out = (const float*)d_in[4];  // [256,512]
  const float* pos_tab  = (const float*)d_in[5];  // [729,8]

  const size_t B = (size_t)in_sizes[0] / (256 * NTOK);   // 128

  // d_out = out [B,256,196] f32  ++  attn [B,8,196,196] f32
  float* outp = (float*)d_out;
  float* attn = outp + B * 256 * NTOK;

  // workspace carving (bf16 intermediates for WMMA feeds; f32 for attn@v out)
  auto al = [](size_t v) { return (v + 255) & ~(size_t)255; };
  char* ws = (char*)d_ws;
  size_t off = 0;
  unsigned short* dwx = (unsigned short*)(ws + off); off = al(off + B * NTOK * 256 * 2);
  unsigned short* q_s = (unsigned short*)(ws + off); off = al(off + B * HEADS * NTOK * DH * 2);
  unsigned short* k_s = (unsigned short*)(ws + off); off = al(off + B * HEADS * NTOK * DH * 2);
  unsigned short* v_t = (unsigned short*)(ws + off); off = al(off + B * HEADS * DH * NTOK * 2);
  float*          o_s = (float*)(ws + off);          off = al(off + B * HEADS * NTOK * DH * 4);
  unsigned short* dwo = (unsigned short*)(ws + off); off = al(off + B * NTOK * 512 * 2);

  int tot1 = (int)(B * 256 * NTOK);
  dw3x3_x_kernel<<<(tot1 + 255) / 256, 256, 0, stream>>>(x, w_dw_qkv, dwx, tot1);

  pw_qkv_kernel<<<(int)(B * 12), 256, 0, stream>>>(w_pw_qkv, dwx, q_s, k_s, v_t);

  attn_scores_softmax_kernel<<<(int)(B * HEADS), 256, 0, stream>>>(q_s, k_s, pos_tab, attn);

  attn_v_kernel<<<(int)(B * HEADS), 256, 0, stream>>>(attn, v_t, o_s);

  int tot2 = (int)(B * 512 * NTOK);
  dw3x3_out_kernel<<<(tot2 + 255) / 256, 256, 0, stream>>>(o_s, w_dw_out, dwo, tot2);

  pw_out_kernel<<<(int)(B * 2), 256, 0, stream>>>(w_pw_out, dwo, outp);
}
